// CausalMessagePassingLayer_90237262889057
// MI455X (gfx1250) — compile-verified, matrix operand order checked by reading, not run on hardware
//
#include <hip/hip_runtime.h>
#include <hip/hip_bf16.h>
#include <stdint.h>

// ---------------------------------------------------------------------------
// Problem dims (fixed by setup_inputs in the reference)
// ---------------------------------------------------------------------------
#define B_  4
#define N_  4096
#define M_  16384
#define EG_ 65536
#define EA_ 65536
#define D_  512
#define DK_ 64

typedef __attribute__((ext_vector_type(16))) __bf16 v16bf;
typedef __attribute__((ext_vector_type(8)))  float  v8f;

union FragU { uint4 q[2]; v16bf v; };

__device__ __forceinline__ unsigned short f2bf(float f) {
  unsigned u = __float_as_uint(f);
  u += 0x7FFFu + ((u >> 16) & 1u);          // round-to-nearest-even
  return (unsigned short)(u >> 16);
}

// ordered-uint encoding for float atomicMax
__device__ __forceinline__ unsigned fenc(float f) {
  unsigned u = __float_as_uint(f);
  return (u & 0x80000000u) ? ~u : (u | 0x80000000u);
}
__device__ __forceinline__ float fdec(unsigned e) {
  unsigned u = (e & 0x80000000u) ? (e & 0x7FFFFFFFu) : ~e;
  return __uint_as_float(u);
}
#define ENC_NEG_INF 0x007FFFFFu   // fenc(-inf)

// ---------------------------------------------------------------------------
// WMMA bf16 GEMM:  C[Mrows,Ncols] = A[Mrows,K] * B[K,Ncols] (+ bias)
// A row-major bf16, B supplied transposed: BT[Ncols,K] bf16, C fp32.
// grid = (Mrows/128, Ncols/64), block = 256 (8 waves).
// Each wave: 4 stacked 16x16 tiles sharing one B fragment per k-step.
// ---------------------------------------------------------------------------
__global__ __launch_bounds__(256, 2) void gemm_bf16_wmma(
    const unsigned short* __restrict__ A,
    const unsigned short* __restrict__ BT,
    const float* __restrict__ bias,      // [Ncols] or nullptr
    float* __restrict__ C,
    int K, int Ncols)
{
  const int lane   = threadIdx.x & 31;
  const int wave   = threadIdx.x >> 5;
  const int l16    = lane & 15;
  const int lhalf  = lane >> 4;
  const int rowBase = blockIdx.x * 128 + (wave >> 2) * 64;
  const int colBase = blockIdx.y * 64  + (wave & 3) * 16;
  const int kOff    = lhalf ? 8 : 0;   // ISA 16-bit A 16x32 lane layout

  const unsigned short* bp  = BT + (size_t)(colBase + l16) * K + kOff;
  const unsigned short* ap0 = A  + (size_t)(rowBase +  0 + l16) * K + kOff;
  const unsigned short* ap1 = ap0 + (size_t)16 * K;
  const unsigned short* ap2 = ap0 + (size_t)32 * K;
  const unsigned short* ap3 = ap0 + (size_t)48 * K;

  v8f acc[4] = {};
  for (int k0 = 0; k0 < K; k0 += 32) {
    FragU bf, a0, a1, a2, a3;
    bf.q[0] = *(const uint4*)(bp  + k0); bf.q[1] = *(const uint4*)(bp  + k0 + 16);
    a0.q[0] = *(const uint4*)(ap0 + k0); a0.q[1] = *(const uint4*)(ap0 + k0 + 16);
    a1.q[0] = *(const uint4*)(ap1 + k0); a1.q[1] = *(const uint4*)(ap1 + k0 + 16);
    a2.q[0] = *(const uint4*)(ap2 + k0); a2.q[1] = *(const uint4*)(ap2 + k0 + 16);
    a3.q[0] = *(const uint4*)(ap3 + k0); a3.q[1] = *(const uint4*)(ap3 + k0 + 16);
    acc[0] = __builtin_amdgcn_wmma_f32_16x16x32_bf16(false, a0.v, false, bf.v, (short)0, acc[0], false, false);
    acc[1] = __builtin_amdgcn_wmma_f32_16x16x32_bf16(false, a1.v, false, bf.v, (short)0, acc[1], false, false);
    acc[2] = __builtin_amdgcn_wmma_f32_16x16x32_bf16(false, a2.v, false, bf.v, (short)0, acc[2], false, false);
    acc[3] = __builtin_amdgcn_wmma_f32_16x16x32_bf16(false, a3.v, false, bf.v, (short)0, acc[3], false, false);
  }

  const float bcol = bias ? bias[colBase + l16] : 0.0f;
  #pragma unroll
  for (int sub = 0; sub < 4; ++sub) {
    const int m = rowBase + sub * 16 + lhalf * 8;   // C layout: VGPR r -> M=r(+8)
    float* cp = C + (size_t)m * Ncols + colBase + l16;
    #pragma unroll
    for (int r = 0; r < 8; ++r) cp[(size_t)r * Ncols] = acc[sub][r] + bcol;
  }
}

// ---------------------------------------------------------------------------
// Elementwise / scatter kernels
// ---------------------------------------------------------------------------
__global__ void fill_f32(float* p, float v, int n) {
  int i = blockIdx.x * 256 + threadIdx.x; if (i < n) p[i] = v;
}
__global__ void fill_u32(unsigned* p, unsigned v, int n) {
  int i = blockIdx.x * 256 + threadIdx.x; if (i < n) p[i] = v;
}

// edge-token gather + convert: A[m,d] = bf16(tok[t2e[m], d]); one block per row
__global__ void gather_edges_bf16(const float* __restrict__ tok,
                                  const int* __restrict__ t2e,
                                  unsigned short* __restrict__ A, int D) {
  const int m = blockIdx.x;
  const size_t src = (size_t)t2e[m] * D;
  for (int d = threadIdx.x; d < D; d += 256)
    A[(size_t)m * D + d] = f2bf(tok[src + d]);
}

// row convert fp32 -> bf16
__global__ void cvt_bf16(const float* __restrict__ src,
                         unsigned short* __restrict__ dst, int D) {
  const size_t r = (size_t)blockIdx.x * D;
  for (int d = threadIdx.x; d < D; d += 256) dst[r + d] = f2bf(src[r + d]);
}

// weight transpose+convert: WT[n,k] = bf16(W[k,n]); one block per output row n
__global__ void wtrans_bf16(const float* __restrict__ W,
                            unsigned short* __restrict__ WT, int K, int Nc) {
  const int n = blockIdx.x;
  for (int k = threadIdx.x; k < K; k += 256)
    WT[(size_t)n * K + k] = f2bf(W[(size_t)k * Nc + n]);
}

__global__ void deg_count(const int* __restrict__ dst, float* __restrict__ deg, int E) {
  int i = blockIdx.x * 256 + threadIdx.x;
  if (i < E) atomicAdd(&deg[dst[i]], 1.0f);
}
__global__ void rsqrt_inplace(float* p, int n) {
  int i = blockIdx.x * 256 + threadIdx.x;
  if (i < n) p[i] = rsqrtf(p[i]);          // deg >= 1 always (self loop)
}

// agg[m,:] = dinv[m]^2 * xw[m,:] + b   (self-loop term + bias, non-atomic init)
__global__ void gcn_init(const float* __restrict__ xw, const float* __restrict__ dinv,
                         const float* __restrict__ b, float* __restrict__ agg, int D) {
  const int m = blockIdx.x;
  const float nrm = dinv[m] * dinv[m];
  const size_t r = (size_t)m * D;
  for (int d = threadIdx.x; d < D; d += 256) agg[r + d] = nrm * xw[r + d] + b[d];
}

// agg[dst,:] += dinv[src]*dinv[dst] * xw[src,:]; one block per graph edge
__global__ void gcn_scatter(const float* __restrict__ xw, const float* __restrict__ dinv,
                            const int* __restrict__ src, const int* __restrict__ dst,
                            float* __restrict__ agg, int D) {
  const int e = blockIdx.x;
  const int s = src[e], d = dst[e];
  const float nrm = dinv[s] * dinv[d];
  const size_t rs = (size_t)s * D, rd = (size_t)d * D;
  for (int dd = threadIdx.x; dd < D; dd += 256)
    atomicAdd(&agg[rd + dd], nrm * xw[rs + dd]);
}

// one wave per attention edge: dots[e] = dot(q[ed],k[es]) / sqrt(DK)
__global__ void attn_dots(const float* __restrict__ q, const float* __restrict__ k,
                          const int* __restrict__ es, const int* __restrict__ ed,
                          float* __restrict__ dots, float inv_scale) {
  const int e = blockIdx.x * 8 + (threadIdx.x >> 5);
  const int lane = threadIdx.x & 31;
  const float2 qa = ((const float2*)(q + (size_t)ed[e] * DK_))[lane];
  const float2 ka = ((const float2*)(k + (size_t)es[e] * DK_))[lane];
  float p = qa.x * ka.x + qa.y * ka.y;
  #pragma unroll
  for (int off = 16; off; off >>= 1) p += __shfl_xor(p, off, 32);
  if (lane == 0) dots[e] = p * inv_scale;
}

__global__ void seg_max(const float* __restrict__ dots, const int* __restrict__ ed,
                        unsigned* __restrict__ mEnc, int E) {
  int i = blockIdx.x * 256 + threadIdx.x;
  if (i < E) atomicMax(&mEnc[ed[i]], fenc(dots[i]));
}
__global__ void seg_decode(const unsigned* __restrict__ mEnc, float* __restrict__ mflt,
                           float* __restrict__ den, int n) {
  int i = blockIdx.x * 256 + threadIdx.x;
  if (i < n) {
    float f = fdec(mEnc[i]);
    mflt[i] = (fabsf(f) < __int_as_float(0x7F800000)) ? f : 0.0f;  // isfinite
    den[i] = 0.0f;
  }
}
__global__ void attn_w(const float* __restrict__ dots, const float* __restrict__ mflt,
                       const int* __restrict__ ed, float* __restrict__ w,
                       float* __restrict__ den, int E) {
  int i = blockIdx.x * 256 + threadIdx.x;
  if (i < E) {
    float v = __expf(dots[i] - mflt[ed[i]]);
    w[i] = v;
    atomicAdd(&den[ed[i]], v);
  }
}
// gatt[ed,:] += (w/den) * values[es,:]; one block per attention edge
__global__ void attn_scatter(const float* __restrict__ w, const float* __restrict__ den,
                             const int* __restrict__ es, const int* __restrict__ ed,
                             const float* __restrict__ vals, float* __restrict__ gatt, int D) {
  const int e = blockIdx.x;
  const int s = es[e], d = ed[e];
  const float dn = den[d];
  const float wn = w[e] / (dn > 0.0f ? dn : 1.0f);
  const size_t rs = (size_t)s * D, rd = (size_t)d * D;
  for (int dd = threadIdx.x; dd < D; dd += 256)
    atomicAdd(&gatt[rd + dd], wn * vals[rs + dd]);
}

// new = t + tanh(ga)*gatt  (fp32 + bf16 copy for the Wl GEMM)
__global__ void make_new(const float* __restrict__ tok, const float* __restrict__ gatt,
                         const float* __restrict__ ga, float* __restrict__ nw,
                         unsigned short* __restrict__ nw_bf, int D) {
  const float g = tanhf(ga[0]);
  const size_t r = (size_t)blockIdx.x * D;
  for (int d = threadIdx.x; d < D; d += 256) {
    float v = tok[r + d] + g * gatt[r + d];
    nw[r + d] = v;
    nw_bf[r + d] = f2bf(v);
  }
}
// out = new + tanh(gb)*lin   (lin already includes bias bl from GEMM epilogue)
__global__ void final_out(const float* __restrict__ nw, const float* __restrict__ lin,
                          const float* __restrict__ gb, float* __restrict__ out, int D) {
  const float g = tanhf(gb[0]);
  const size_t r = (size_t)blockIdx.x * D;
  for (int d = threadIdx.x; d < D; d += 256)
    out[r + d] = nw[r + d] + g * lin[r + d];
}

// ---------------------------------------------------------------------------
// Host orchestration
// ---------------------------------------------------------------------------
extern "C" void kernel_launch(void* const* d_in, const int* in_sizes, int n_in,
                              void* d_out, int out_size, void* d_ws, size_t ws_size,
                              hipStream_t stream) {
  (void)in_sizes; (void)n_in; (void)out_size; (void)ws_size;
  const float* tok  = (const float*)d_in[0];     // [B,N,D]
  const int*   t2e  = (const int*)d_in[1];       // [B,M]
  const int*   ei   = (const int*)d_in[2];       // [B,2,Eg]
  const int*   es   = (const int*)d_in[3];       // [B,Ea]
  const int*   ed   = (const int*)d_in[4];       // [B,Ea]
  const float* Wg   = (const float*)d_in[5];     // [D,D]
  const float* bg   = (const float*)d_in[6];     // [D]
  const float* Wk   = (const float*)d_in[7];     // [D,DK]
  const float* bk   = (const float*)d_in[8];     // [DK]
  const float* Wq   = (const float*)d_in[9];     // [D,DK]
  const float* bq   = (const float*)d_in[10];    // [DK]
  const float* Wl   = (const float*)d_in[11];    // [D,D]
  const float* bl   = (const float*)d_in[12];    // [D]
  const float* ga   = (const float*)d_in[13];    // [1]
  const float* gb   = (const float*)d_in[14];    // [1]
  float* out = (float*)d_out;                    // [B,N,D]

  // ---- carve workspace ----
  size_t off = 0;
  char* wsb = (char*)d_ws;
  auto carve = [&](size_t bytes) -> void* {
    void* p = wsb + off; off += (bytes + 255) & ~(size_t)255; return p;
  };
  unsigned short* WgT   = (unsigned short*)carve((size_t)D_ * D_ * 2);
  unsigned short* WkT   = (unsigned short*)carve((size_t)DK_ * D_ * 2);
  unsigned short* WqT   = (unsigned short*)carve((size_t)DK_ * D_ * 2);
  unsigned short* WlT   = (unsigned short*)carve((size_t)D_ * D_ * 2);
  unsigned short* Abf   = (unsigned short*)carve((size_t)M_ * D_ * 2);  // gather / gcn-out bf16
  unsigned short* Tbf   = (unsigned short*)carve((size_t)N_ * D_ * 2);  // t bf16 / new bf16
  float* xw   = (float*)carve((size_t)M_ * D_ * 4);
  float* agg  = (float*)carve((size_t)M_ * D_ * 4);                     // GCN out = values
  float* kbuf = (float*)carve((size_t)M_ * DK_ * 4);
  float* qbuf = (float*)carve((size_t)N_ * DK_ * 4);
  float* dots = (float*)carve((size_t)EA_ * 4);
  float* wbuf = (float*)carve((size_t)EA_ * 4);
  unsigned* mEnc = (unsigned*)carve((size_t)N_ * 4);
  float* mflt = (float*)carve((size_t)N_ * 4);
  float* den  = (float*)carve((size_t)N_ * 4);
  float* dinv = (float*)carve((size_t)M_ * 4);
  float* gatt = (float*)carve((size_t)N_ * D_ * 4);
  float* newb = (float*)carve((size_t)N_ * D_ * 4);
  float* lin  = (float*)carve((size_t)N_ * D_ * 4);

  // ---- one-time weight transposes (batch independent) ----
  wtrans_bf16<<<D_,  256, 0, stream>>>(Wg, WgT, D_, D_);
  wtrans_bf16<<<DK_, 256, 0, stream>>>(Wk, WkT, D_, DK_);
  wtrans_bf16<<<DK_, 256, 0, stream>>>(Wq, WqT, D_, DK_);
  wtrans_bf16<<<D_,  256, 0, stream>>>(Wl, WlT, D_, D_);

  for (int b = 0; b < B_; ++b) {
    const float* tokb = tok + (size_t)b * N_ * D_;
    const int* t2eb = t2e + (size_t)b * M_;
    const int* srcb = ei + (size_t)b * 2 * EG_;
    const int* dstb = srcb + EG_;
    const int* esb  = es + (size_t)b * EA_;
    const int* edb  = ed + (size_t)b * EA_;
    float* outb = out + (size_t)b * N_ * D_;

    // 1) gather edge tokens (bf16) ; xw = gather @ W_gnn
    gather_edges_bf16<<<M_, 256, 0, stream>>>(tokb, t2eb, Abf, D_);
    gemm_bf16_wmma<<<dim3(M_/128, D_/64), 256, 0, stream>>>(Abf, WgT, nullptr, xw, D_, D_);

    // 2) GCN normalization + aggregation
    fill_f32<<<(M_+255)/256, 256, 0, stream>>>(dinv, 1.0f, M_);          // self loops
    deg_count<<<(EG_+255)/256, 256, 0, stream>>>(dstb, dinv, EG_);
    rsqrt_inplace<<<(M_+255)/256, 256, 0, stream>>>(dinv, M_);
    gcn_init<<<M_, 256, 0, stream>>>(xw, dinv, bg, agg, D_);
    gcn_scatter<<<EG_, 256, 0, stream>>>(xw, dinv, srcb, dstb, agg, D_);

    // 3) k / q projections
    cvt_bf16<<<M_, 256, 0, stream>>>(agg, Abf, D_);
    gemm_bf16_wmma<<<dim3(M_/128, 1), 256, 0, stream>>>(Abf, WkT, bk, kbuf, D_, DK_);
    cvt_bf16<<<N_, 256, 0, stream>>>(tokb, Tbf, D_);
    gemm_bf16_wmma<<<dim3(N_/128, 1), 256, 0, stream>>>(Tbf, WqT, bq, qbuf, D_, DK_);

    // 4) segment-softmax cross attention
    attn_dots<<<EA_/8, 256, 0, stream>>>(qbuf, kbuf, esb, edb, dots, 0.125f);
    fill_u32<<<(N_+255)/256, 256, 0, stream>>>(mEnc, ENC_NEG_INF, N_);
    seg_max<<<(EA_+255)/256, 256, 0, stream>>>(dots, edb, mEnc, EA_);
    seg_decode<<<(N_+255)/256, 256, 0, stream>>>(mEnc, mflt, den, N_);
    attn_w<<<(EA_+255)/256, 256, 0, stream>>>(dots, mflt, edb, wbuf, den, EA_);
    fill_f32<<<((N_*D_)+255)/256, 256, 0, stream>>>(gatt, 0.0f, N_*D_);
    attn_scatter<<<EA_, 256, 0, stream>>>(wbuf, den, esb, edb, agg, gatt, D_);

    // 5) residual + gated linear
    make_new<<<N_, 256, 0, stream>>>(tokb, gatt, ga, newb, Tbf, D_);
    gemm_bf16_wmma<<<dim3(N_/128, D_/64), 256, 0, stream>>>(Tbf, WlT, bl, lin, D_, D_);
    final_out<<<N_, 256, 0, stream>>>(newb, lin, gb, outb, D_);
  }
}
